// CrossAttention_20409684591353
// MI455X (gfx1250) — matrix operand order repaired and checked by reading, not compiled, not run on hardware
//
#include <hip/hip_runtime.h>
#include <hip/hip_bf16.h>

#define DEVI __device__ __forceinline__

typedef __bf16 bf16x16 __attribute__((ext_vector_type(16)));
typedef float  f32x8   __attribute__((ext_vector_type(8)));
typedef unsigned int u32x4 __attribute__((ext_vector_type(4)));
typedef int          i32x4 __attribute__((ext_vector_type(4)));
typedef int          i32x8 __attribute__((ext_vector_type(8)));

static constexpr int   B_  = 4, S_ = 1024, C_ = 1024, H_ = 16, HD_ = 64;
static constexpr int   M_  = B_ * S_;          // 4096 rows for all big GEMMs
static constexpr float SCALE_ = 0.125f;        // HD^-0.5

// ---- CDNA5 async / TDM feature detection (device pass only) ---------------
#if defined(__AMDGCN__) && __has_builtin(__builtin_amdgcn_global_load_async_to_lds_b128) && __has_builtin(__builtin_amdgcn_s_wait_asynccnt)
#define HAS_ASYNC 1
#else
#define HAS_ASYNC 0
#endif
#if defined(__AMDGCN__) && __has_builtin(__builtin_amdgcn_tensor_load_to_lds) && __has_builtin(__builtin_amdgcn_s_wait_tensorcnt)
#define HAS_TDM 1
#else
#define HAS_TDM 0
#endif

union FragA { unsigned int u[8]; bf16x16 v; };

DEVI unsigned short f2bf(float f) {
  unsigned int u = __float_as_uint(f);
  u += 0x7FFFu + ((u >> 16) & 1u);             // round-to-nearest-even
  return (unsigned short)(u >> 16);
}

#if HAS_ASYNC
// 16B global -> LDS async DMA (ASYNCcnt). Builtin takes typed v4i pointers:
// (v4i addrspace(1)* src, v4i addrspace(3)* dst, imm offset, imm cpol).
DEVI void async_copy_b128(const void* g, void* l) {
  typedef int v4i __attribute__((vector_size(16)));
  typedef __attribute__((address_space(1))) v4i* gp_t;
  typedef __attribute__((address_space(3))) v4i* lp_t;
  __builtin_amdgcn_global_load_async_to_lds_b128(
      (gp_t)(unsigned long long)g,
      (lp_t)(unsigned int)(unsigned long long)l, 0, 0);
}
#endif

#if HAS_TDM
// Issue a 2D TDM tile load: tile_d0 dwords x tile_d1 rows out of a tensor with
// row stride (dwords), data_size = 4B. Descriptors per ISA 08_async_tensor §8.
DEVI void tdm_load_2d(const void* gtile, void* ldst, int tile_d0, int tile_d1,
                      int tensor_d0, int tensor_d1, int stride_dw) {
  unsigned long long ga = (unsigned long long)gtile;
  unsigned int lo = (unsigned int)(unsigned long long)ldst;  // LDS byte addr
  u32x4 g0 = {1u,                                   // count=1, user mode
              lo,                                   // lds_addr [63:32]
              (unsigned int)ga,                     // global_addr lo
              (unsigned int)((ga >> 32) & 0x1FFFFFFu) | 0x80000000u}; // +type=2
  i32x8 g1 = {(int)(2u << 16),                      // data_size=4B
              (int)((unsigned)tensor_d0 << 16),     // tensor_dim0 [79:48] lo16
              (int)(((unsigned)tensor_d0 >> 16) | ((unsigned)tensor_d1 << 16)),
              (int)(((unsigned)tensor_d1 >> 16) | ((unsigned)tile_d0 << 16)),
              (int)(unsigned)tile_d1,               // tile_dim1 [143:128]
              (int)(unsigned)stride_dw,             // tensor_dim0_stride lo32
              0, 0};
  i32x4 z4 = {0, 0, 0, 0};
#if __clang_major__ >= 23
  i32x8 z8 = {0, 0, 0, 0, 0, 0, 0, 0};
  __builtin_amdgcn_tensor_load_to_lds(g0, g1, z4, z4, z8, 0);
#else
  __builtin_amdgcn_tensor_load_to_lds(g0, g1, z4, z4, 0);
#endif
}
#endif

// ---------------------------------------------------------------------------
// fp32 -> bf16 cast (weights and activations)
// ---------------------------------------------------------------------------
__global__ void cast_f32_bf16(const float* __restrict__ src,
                              unsigned short* __restrict__ dst, int n) {
  for (int i = blockIdx.x * blockDim.x + threadIdx.x; i < n;
       i += gridDim.x * blockDim.x)
    dst[i] = f2bf(src[i]);
}

// ---------------------------------------------------------------------------
// GEMM: Y = epilogue( X[4096,1024](bf16) @ W[1024,1024](bf16) + bias )
// block tile 128x128, BK=32, 8 waves each computing 32x64 via 2x4 WMMA tiles
// A tile staged by async-to-LDS DMA; B tile sync-staged transposed in LDS.
// modes: 0: Ybf = bf16(val*alpha)
//        1: Ybf = bf16(val), Yf = val
//        2: Ybf = bf16(gelu_exact(val))
//        3: Yf  = val + res   (final residual, fp32 out)
// ---------------------------------------------------------------------------
__global__ __launch_bounds__(256)
void gemm_bf16_wmma(const unsigned short* __restrict__ X,
                    const unsigned short* __restrict__ W,
                    const float* __restrict__ bias,
                    float alpha, int mode,
                    unsigned short* __restrict__ Ybf,
                    float* __restrict__ Yf,
                    const float* __restrict__ res) {
  __shared__ unsigned int Abuf[128 * 16];  // [row][kpair]  128x32 bf16 = 8KB
  __shared__ unsigned int BtU [128 * 16];  // [n][kpair]    transposed W tile

  const int tid  = threadIdx.x;
  const int wave = tid >> 5, lane = tid & 31;
  const int half = lane >> 4, ln = lane & 15;
  const int wm = wave >> 1, wn = wave & 1;
  const int n0 = blockIdx.x * 128, m0 = blockIdx.y * 128;

  const unsigned int* Xu = (const unsigned int*)X;
  const unsigned int* Wu = (const unsigned int*)W;

  f32x8 acc[2][4];
  for (int mt = 0; mt < 2; ++mt)
    for (int nt = 0; nt < 4; ++nt)
      for (int v = 0; v < 8; ++v) acc[mt][nt][v] = 0.0f;

  for (int kb = 0; kb < 1024; kb += 32) {
    __syncthreads();
#if HAS_ASYNC
    // A tile: 512 x 16B chunks direct global->LDS DMA (ASYNCcnt)
    for (int i = 0; i < 2; ++i) {
      int idx = tid * 2 + i;
      int row = idx >> 2, c16 = idx & 3;
      const char* g = (const char*)X + (long)(m0 + row) * 2048 + kb * 2 + c16 * 16;
      async_copy_b128(g, (char*)Abuf + idx * 16);
    }
#else
    for (int i = 0; i < 8; ++i) {
      int idx = tid * 8 + i;
      int row = idx >> 4, kp = idx & 15;
      Abuf[idx] = Xu[(m0 + row) * 512 + (kb >> 1) + kp];
    }
#endif
    // B tile: sync load + in-LDS transpose (DMA cannot transpose elements)
    for (int i = 0; i < 8; ++i) {
      int idx = tid * 8 + i;
      int k = idx >> 6, np = idx & 63;
      unsigned int u = Wu[(kb + k) * 512 + (n0 >> 1) + np];
      ((unsigned short*)BtU)[(2 * np    ) * 32 + k] = (unsigned short)(u & 0xFFFFu);
      ((unsigned short*)BtU)[(2 * np + 1) * 32 + k] = (unsigned short)(u >> 16);
    }
    if (kb + 32 < 1024) {  // global_prefetch_b8 for next tile
      __builtin_prefetch(&Xu[(m0 + (tid >> 1)) * 512 + ((kb + 32) >> 1)], 0, 1);
      __builtin_prefetch(&Wu[(kb + 32 + (tid >> 3)) * 512 + (n0 >> 1)], 0, 1);
    }
#if HAS_ASYNC
    __builtin_amdgcn_s_wait_asynccnt(0);
#endif
    __syncthreads();

    FragA a[2];
    for (int mt = 0; mt < 2; ++mt) {
      int row = wm * 32 + 16 * mt + ln;
      for (int j = 0; j < 8; ++j) {
        int kp = (j < 4) ? (4 * half + j) : (8 + 4 * half + (j - 4));
        a[mt].u[j] = Abuf[row * 16 + kp];
      }
    }
    for (int nt = 0; nt < 4; ++nt) {
      FragA b;
      int n = wn * 64 + 16 * nt + ln;
      for (int j = 0; j < 8; ++j) b.u[j] = BtU[n * 16 + ((j < 4) ? (4 * half + j) : (8 + 4 * half + (j - 4)))];
      for (int mt = 0; mt < 2; ++mt)
        acc[mt][nt] = __builtin_amdgcn_wmma_f32_16x16x32_bf16(
            false, a[mt].v, false, b.v, (short)0, acc[mt][nt], false, false);
    }
  }

  for (int mt = 0; mt < 2; ++mt)
    for (int nt = 0; nt < 4; ++nt) {
      int colb = n0 + wn * 64 + 16 * nt + ln;
      float bv = bias[colb];
      for (int v = 0; v < 8; ++v) {
        int row = m0 + wm * 32 + 16 * mt + 8 * half + v;
        float val = acc[mt][nt][v] + bv;
        long idx = (long)row * 1024 + colb;
        if (mode == 0) {
          Ybf[idx] = f2bf(val * alpha);
        } else if (mode == 1) {
          Ybf[idx] = f2bf(val); Yf[idx] = val;
        } else if (mode == 2) {
          float g = 0.5f * val * (1.0f + erff(val * 0.70710678118f));
          Ybf[idx] = f2bf(g);
        } else {
          Yf[idx] = val + res[idx];
        }
      }
    }
}

// ---------------------------------------------------------------------------
// Flash attention: one block = one (b,h) * 128 query rows, 8 waves x 16 rows.
// K/V chunks staged by the Tensor Data Mover (TENSORcnt); online softmax per
// wave; WMMA for QK^T and PV; mask from int32 inputs.
// ---------------------------------------------------------------------------
__global__ __launch_bounds__(256)
void flash_attn(const unsigned short* __restrict__ Q,
                const unsigned short* __restrict__ K,
                const unsigned short* __restrict__ V,
                const int* __restrict__ mask,
                unsigned short* __restrict__ O) {
  __shared__ unsigned int Klds[64 * 32];       // 64 keys x 64 dims bf16
  __shared__ unsigned int Vlds[64 * 32];
  __shared__ unsigned short Plds[8 * 16 * 64]; // per-wave P tile bounce

  const int tid  = threadIdx.x;
  const int wave = tid >> 5, lane = tid & 31;
  const int half = lane >> 4, ln = lane & 15;
  const int b = blockIdx.x >> 4, h = blockIdx.x & 15;
  const int q0 = blockIdx.y * 128 + wave * 16;

  const unsigned int* Qu = (const unsigned int*)Q;
  const unsigned int* PldsU = (const unsigned int*)Plds;
  const unsigned short* V16 = (const unsigned short*)Vlds;

  FragA aq[2];                                  // Q rows as A-frags, K split 2x32
  for (int s = 0; s < 2; ++s)
    for (int j = 0; j < 8; ++j) {
      int koff = 32 * s + ((j < 4) ? (8 * half + 2 * j)
                                   : (16 + 8 * half + 2 * (j - 4)));
      aq[s].u[j] = Qu[(long)(b * S_ + q0 + ln) * 512 + h * 32 + (koff >> 1)];
    }

  float mrow[8], ssum[8];
  f32x8 acco[4];
  for (int v = 0; v < 8; ++v) { mrow[v] = -__builtin_inff(); ssum[v] = 0.0f; }
  for (int t = 0; t < 4; ++t)
    for (int v = 0; v < 8; ++v) acco[t][v] = 0.0f;

  for (int kv0 = 0; kv0 < S_; kv0 += 64) {
    __syncthreads();
#if HAS_TDM
    if (wave == 0) {  // one TDM descriptor per chunk; EXEC ignored by TDM
      const char* gK = (const char*)K + (long)(b * S_ + kv0) * 2048 + h * 128;
      const char* gV = (const char*)V + (long)(b * S_ + kv0) * 2048 + h * 128;
      tdm_load_2d(gK, Klds, /*tile_d0=*/32, /*tile_d1=*/64,
                  /*tensor_d0=*/512, /*tensor_d1=*/M_, /*stride_dw=*/512);
      tdm_load_2d(gV, Vlds, 32, 64, 512, M_, 512);
      __builtin_amdgcn_s_wait_tensorcnt(0);
    }
#elif HAS_ASYNC
    for (int i = 0; i < 2; ++i) {
      int idx = tid * 2 + i;                    // 512 x 16B per buffer
      int key = idx >> 3, o16 = idx & 7;
      long gb = (long)(b * S_ + kv0 + key) * 2048 + h * 128 + o16 * 16;
      async_copy_b128((const char*)K + gb, (char*)Klds + idx * 16);
      async_copy_b128((const char*)V + gb, (char*)Vlds + idx * 16);
    }
    __builtin_amdgcn_s_wait_asynccnt(0);
#else
    for (int i = 0; i < 8; ++i) {
      int idx = tid * 8 + i;
      int key = idx >> 5, dp = idx & 31;
      long g = (long)(b * S_ + kv0 + key) * 512 + h * 32 + dp;
      Klds[idx] = ((const unsigned int*)K)[g];
      Vlds[idx] = ((const unsigned int*)V)[g];
    }
#endif
    __syncthreads();

    int mk[4];
    for (int t = 0; t < 4; ++t) {
      int col = kv0 + 16 * t + ln;
      mk[t] = (mask[b * 3 * S_ + col] | mask[b * 3 * S_ + S_ + col]) != 0;
    }

    float sv[4][8];
    for (int t = 0; t < 4; ++t) {               // S = Q @ K^T (8 WMMAs/chunk)
      f32x8 accs;
      for (int v = 0; v < 8; ++v) accs[v] = 0.0f;
      for (int s = 0; s < 2; ++s) {
        FragA bk;
        for (int j = 0; j < 8; ++j)
          bk.u[j] = Klds[(16 * t + ln) * 32 + 16 * s + ((j < 4) ? (4 * half + j) : (8 + 4 * half + (j - 4)))];
        accs = __builtin_amdgcn_wmma_f32_16x16x32_bf16(
            false, aq[s].v, false, bk.v, (short)0, accs, false, false);
      }
      for (int v = 0; v < 8; ++v)
        sv[t][v] = mk[t] ? accs[v] : -__builtin_inff();
    }

    float mnew[8];
    for (int v = 0; v < 8; ++v) {               // row max over N (16-lane group)
      float cm = sv[0][v];
      for (int t = 1; t < 4; ++t) cm = fmaxf(cm, sv[t][v]);
      for (int d = 1; d < 16; d <<= 1) cm = fmaxf(cm, __shfl_xor(cm, d, 32));
      mnew[v] = fmaxf(mrow[v], cm);
    }

    float fac[8], mg[8];
    for (int v = 0; v < 8; ++v) {
      mg[v]  = (mnew[v] == -__builtin_inff()) ? 0.0f : mnew[v];
      fac[v] = (mrow[v] == -__builtin_inff()) ? 0.0f : __expf(mrow[v] - mnew[v]);
      mrow[v] = mnew[v];
    }

    float prow[8];
    for (int v = 0; v < 8; ++v) prow[v] = 0.0f;
    for (int t = 0; t < 4; ++t)
      for (int v = 0; v < 8; ++v) {
        float p = __expf(sv[t][v] - mg[v]);     // masked: exp(-inf)=0
        sv[t][v] = p;
        prow[v] += p;
      }
    for (int v = 0; v < 8; ++v) {
      float ps = prow[v];
      for (int d = 1; d < 16; d <<= 1) ps += __shfl_xor(ps, d, 32);
      ssum[v] = ssum[v] * fac[v] + ps;
    }
    for (int t = 0; t < 4; ++t)
      for (int v = 0; v < 8; ++v) acco[t][v] *= fac[v];

    // bounce P through LDS to convert C-layout -> A-layout (bf16)
    for (int t = 0; t < 4; ++t)
      for (int v = 0; v < 8; ++v)
        Plds[wave * 1024 + (8 * half + v) * 64 + 16 * t + ln] = f2bf(sv[t][v]);
    __syncthreads();

    for (int s = 0; s < 2; ++s) {               // O += P @ V (8 WMMAs/chunk)
      FragA pa;
      for (int j = 0; j < 8; ++j) {
        int koff = 32 * s + ((j < 4) ? (8 * half + 2 * j)
                                     : (16 + 8 * half + 2 * (j - 4)));
        pa.u[j] = PldsU[wave * 512 + ln * 32 + (koff >> 1)];
      }
      for (int t = 0; t < 4; ++t) {
        FragA vb;                               // gather-transpose V from LDS
        for (int j = 0; j < 8; ++j) {
          int k = 32 * s + ((j < 4) ? (8 * half + 2 * j) : (16 + 8 * half + 2 * (j - 4)));
          int n = 16 * t + ln;
          unsigned int lo = V16[k * 64 + n];
          unsigned int hi = V16[(k + 1) * 64 + n];
          vb.u[j] = lo | (hi << 16);
        }
        acco[t] = __builtin_amdgcn_wmma_f32_16x16x32_bf16(
            false, pa.v, false, vb.v, (short)0, acco[t], false, false);
      }
    }
  }

  float rcp[8];
  for (int v = 0; v < 8; ++v)
    rcp[v] = (ssum[v] > 0.0f) ? (1.0f / ssum[v]) : 0.0f;  // fully-masked rows -> 0
  for (int t = 0; t < 4; ++t)
    for (int v = 0; v < 8; ++v) {
      int row = q0 + 8 * half + v;
      int d = 16 * t + ln;
      O[(long)(b * S_ + row) * C_ + h * 64 + d] = f2bf(acco[t][v] * rcp[v]);
    }
}

// ---------------------------------------------------------------------------
extern "C" void kernel_launch(void* const* d_in, const int* in_sizes, int n_in,
                              void* d_out, int out_size, void* d_ws,
                              size_t ws_size, hipStream_t stream) {
  (void)in_sizes; (void)n_in; (void)out_size; (void)ws_size;
  const float* query = (const float*)d_in[0];
  const float* key_t = (const float*)d_in[1];
  const float* value = (const float*)d_in[2];
  const int*   mask  = (const int*)d_in[3];
  const float* Wq = (const float*)d_in[4];  const float* bq = (const float*)d_in[5];
  const float* Wk = (const float*)d_in[6];  const float* bk = (const float*)d_in[7];
  const float* Wv = (const float*)d_in[8];  const float* bv = (const float*)d_in[9];
  const float* Wp = (const float*)d_in[10]; const float* bp = (const float*)d_in[11];
  const float* W1 = (const float*)d_in[12]; const float* b1 = (const float*)d_in[13];
  const float* W2 = (const float*)d_in[14]; const float* b2 = (const float*)d_in[15];
  float* out = (float*)d_out;

  char* ws = (char*)d_ws;
  size_t off = 0;
  auto alloc = [&](size_t bytes) {
    void* p = ws + off;
    off += (bytes + 255) & ~(size_t)255;
    return p;
  };
  const size_t WB = (size_t)C_ * C_ * 2;   // one bf16 weight matrix
  const size_t AB = (size_t)M_ * C_ * 2;   // one bf16 activation matrix
  unsigned short* Wqb = (unsigned short*)alloc(WB);
  unsigned short* Wkb = (unsigned short*)alloc(WB);
  unsigned short* Wvb = (unsigned short*)alloc(WB);
  unsigned short* Wpb = (unsigned short*)alloc(WB);
  unsigned short* W1b = (unsigned short*)alloc(WB);
  unsigned short* W2b = (unsigned short*)alloc(WB);
  unsigned short* Xqb = (unsigned short*)alloc(AB);
  unsigned short* Xkb = (unsigned short*)alloc(AB);
  unsigned short* Xvb = (unsigned short*)alloc(AB);
  unsigned short* Qp  = (unsigned short*)alloc(AB);
  unsigned short* Kp  = (unsigned short*)alloc(AB);
  unsigned short* Vp  = (unsigned short*)alloc(AB);
  unsigned short* AO  = (unsigned short*)alloc(AB);
  unsigned short* XpB = (unsigned short*)alloc(AB);
  float*          XpF = (float*)alloc((size_t)M_ * C_ * 4);
  unsigned short* Hh  = (unsigned short*)alloc(AB);

  const int nW = C_ * C_, nA = M_ * C_;
  cast_f32_bf16<<<512, 256, 0, stream>>>(Wq, Wqb, nW);
  cast_f32_bf16<<<512, 256, 0, stream>>>(Wk, Wkb, nW);
  cast_f32_bf16<<<512, 256, 0, stream>>>(Wv, Wvb, nW);
  cast_f32_bf16<<<512, 256, 0, stream>>>(Wp, Wpb, nW);
  cast_f32_bf16<<<512, 256, 0, stream>>>(W1, W1b, nW);
  cast_f32_bf16<<<512, 256, 0, stream>>>(W2, W2b, nW);
  cast_f32_bf16<<<1024, 256, 0, stream>>>(query, Xqb, nA);
  cast_f32_bf16<<<1024, 256, 0, stream>>>(key_t, Xkb, nA);
  cast_f32_bf16<<<1024, 256, 0, stream>>>(value, Xvb, nA);

  dim3 ggrid(8, 32), gblk(256);
  // Q/K/V projections (scale folded into Q epilogue, as in reference)
  gemm_bf16_wmma<<<ggrid, gblk, 0, stream>>>(Xqb, Wqb, bq, SCALE_, 0, Qp, nullptr, nullptr);
  gemm_bf16_wmma<<<ggrid, gblk, 0, stream>>>(Xkb, Wkb, bk, 1.0f,   0, Kp, nullptr, nullptr);
  gemm_bf16_wmma<<<ggrid, gblk, 0, stream>>>(Xvb, Wvb, bv, 1.0f,   0, Vp, nullptr, nullptr);

  dim3 agrid(B_ * H_, S_ / 128);
  flash_attn<<<agrid, gblk, 0, stream>>>(Qp, Kp, Vp, mask, AO);

  // output projection (keep fp32 copy for residual), MLP with exact GELU,
  // final residual add into fp32 d_out
  gemm_bf16_wmma<<<ggrid, gblk, 0, stream>>>(AO,  Wpb, bp, 1.0f, 1, XpB, XpF, nullptr);
  gemm_bf16_wmma<<<ggrid, gblk, 0, stream>>>(XpB, W1b, b1, 1.0f, 2, Hh, nullptr, nullptr);
  gemm_bf16_wmma<<<ggrid, gblk, 0, stream>>>(Hh,  W2b, b2, 1.0f, 3, nullptr, out, XpF);
}
